// FixedChargeBaselinedMACE_13872744366305
// MI455X (gfx1250) — matrix operand order repaired and checked by reading, not compiled, not run on hardware
//
#include <hip/hip_runtime.h>
#include <hip/hip_bf16.h>

typedef __attribute__((ext_vector_type(16))) _Float16 v16h;
typedef __attribute__((ext_vector_type(8)))  _Float16 v8h;
typedef __attribute__((ext_vector_type(8)))  float    v8f;
typedef __attribute__((ext_vector_type(4)))  float    v4f;

#define NN   8000
#define EE   100000
#define CC   64
#define NEL  10
#define GG   16
#define LSH  16
#define NBAS 8
#define HR   16
#define NPG  (NN / GG)          // 500 nodes per graph
#define PI_F 3.14159265358979f

// fast silu: x * v_rcp_f32(1 + exp(-x))
__device__ __forceinline__ float siluf(float x) {
    return x * __builtin_amdgcn_rcpf(1.0f + __expf(-x));
}

__device__ __forceinline__ v8f wmma_f16(const v16h& a, const v16h& b, const v8f& c) {
    return __builtin_amdgcn_wmma_f32_16x16x32_f16(
        false, a, false, b, (short)0, c, false, false);
}

// Load an A fragment (16x32 f16 tile, this wave's lane portion) from an LDS
// [16 rows x 64 K] buffer: two contiguous 8-half runs -> two 16B ds loads.
__device__ __forceinline__ v16h ldsA(const _Float16* __restrict__ in,
                                     int col, bool hi, int kc) {
    const v8h lo = *(const v8h*)(in + col * 64 + kc * 32 + (hi ? 16 : 0));
    const v8h hh = *(const v8h*)(in + col * 64 + kc * 32 + (hi ? 24 : 8));
    return __builtin_shufflevector(lo, hh, 0, 1, 2, 3, 4, 5, 6, 7,
                                            8, 9, 10, 11, 12, 13, 14, 15);
}

// ------------------------------------------------------ weight pre-pack (f16)
// B-register layout for a [64(K) x 64(N)] weight: P[((kc*4+nt)*32+lane)*16+e]
__global__ void k_pack_b(const float* __restrict__ W, _Float16* __restrict__ P) {
    int i = blockIdx.x * blockDim.x + threadIdx.x;   // 4096
    if (i >= 4096) return;
    int e = i & 15, lane = (i >> 4) & 31, nt = (i >> 9) & 3, kc = (i >> 11) & 1;
    int col = lane & 15, koff = (lane >= 16) ? 16 : 0;
    P[i] = (_Float16)W[(kc * 32 + koff + e) * 64 + nt * 16 + col];
}

// B-register layout for the [8 x 64] first stage, K zero-padded to 32.
__global__ void k_pack_b0(const float* __restrict__ W, _Float16* __restrict__ P) {
    int i = blockIdx.x * blockDim.x + threadIdx.x;   // 2048
    if (i >= 2048) return;
    int e = i & 15, lane = (i >> 4) & 31, nt = (i >> 9) & 3;
    int col = lane & 15;
    float v = (lane < 16 && e < 8) ? W[e * 64 + nt * 16 + col] : 0.0f;
    P[i] = (_Float16)v;
}

// A-register layout for W^T ([64(c) x 64(d)] weight used as A with M=d, K=c):
// P[((dt*2+kc)*32+lane)*16+e]
__global__ void k_pack_a(const float* __restrict__ W, _Float16* __restrict__ P) {
    int i = blockIdx.x * blockDim.x + threadIdx.x;   // 4096
    if (i >= 4096) return;
    int e = i & 15, lane = (i >> 4) & 31, kc = (i >> 9) & 1, dt = (i >> 10) & 3;
    bool hi = lane >= 16;
    int K = kc * 32 + (hi ? 16 : 0) + e;
    P[i] = (_Float16)W[K * 64 + dt * 16 + (lane & 15)];
}

// ---------------------------------------------------------------- zero output
__global__ void k_zero_out(float* out) {
    if (threadIdx.x < GG) out[threadIdx.x] = 0.0f;
}

// ------------------------------------------------- feats init: feats[:,:,0]=attrs@W_embed
__global__ void k_feats_init(const float* __restrict__ attrs,
                             const float* __restrict__ Wemb,
                             float* __restrict__ feats) {
    int idx = blockIdx.x * blockDim.x + threadIdx.x;   // n*1024 + c*16 + k
    if (idx >= NN * CC * LSH) return;
    int k = idx & 15;
    int c = (idx >> 4) & 63;
    int n = idx >> 10;
    float v = 0.0f;
    if (k == 0) {
        for (int t = 0; t < NEL; ++t) v += attrs[n * NEL + t] * Wemb[t * CC + c];
    }
    feats[idx] = v;
}

// ---------------------------------------------------- per-edge geometry: sh + radial
__global__ void k_edge_geom(const float* __restrict__ pos,
                            const float* __restrict__ shifts,
                            const int* __restrict__ eidx,
                            float* __restrict__ sh,   // [E,16]
                            float* __restrict__ ef) { // [E,8]
    int e = blockIdx.x * blockDim.x + threadIdx.x;
    if (e >= EE) return;
    int snd = eidx[e], rcv = eidx[EE + e];
    float vx = pos[rcv * 3 + 0] - pos[snd * 3 + 0] + shifts[e * 3 + 0];
    float vy = pos[rcv * 3 + 1] - pos[snd * 3 + 1] + shifts[e * 3 + 1];
    float vz = pos[rcv * 3 + 2] - pos[snd * 3 + 2] + shifts[e * 3 + 2];
    float r  = sqrtf(vx * vx + vy * vy + vz * vz);
    float ri = 1.0f / fmaxf(r, 1e-9f);
    float x = vx * ri, y = vy * ri, z = vz * ri;

    const float s3  = 1.7320508075688772f;
    const float s15 = 3.872983346207417f;
    const float s5  = 2.23606797749979f;
    const float c33 = 2.091650066335189f;
    const float c32 = 10.246950765959598f;
    const float c31 = 1.6201851746019651f;
    const float c30 = 1.3228756555322954f;

    float* Y = sh + (size_t)e * 16;
    Y[0]  = 1.0f;
    Y[1]  = s3 * x;
    Y[2]  = s3 * y;
    Y[3]  = s3 * z;
    Y[4]  = s15 * x * y;
    Y[5]  = s15 * y * z;
    Y[6]  = 0.5f * s5 * (3.0f * z * z - 1.0f);
    Y[7]  = s15 * x * z;
    Y[8]  = 0.5f * s15 * (x * x - y * y);
    Y[9]  = c33 * y * (3.0f * x * x - y * y);
    Y[10] = c32 * x * y * z;
    Y[11] = c31 * y * (5.0f * z * z - 1.0f);
    Y[12] = c30 * (5.0f * z * z * z - 3.0f * z);
    Y[13] = c31 * x * (5.0f * z * z - 1.0f);
    Y[14] = 0.5f * c32 * z * (x * x - y * y);
    Y[15] = c33 * x * (x * x - 3.0f * y * y);

    float u = r * (1.0f / 5.0f);
    float u3 = u * u * u;
    float u6 = u3 * u3;
    float cut = (u < 1.0f) ? (1.0f - 28.0f * u6 + 48.0f * u6 * u - 21.0f * u6 * u * u)
                           : 0.0f;
    float pref = 0.6324555320336759f * ri * cut;
    float* F = ef + (size_t)e * 8;
#pragma unroll
    for (int n = 1; n <= NBAS; ++n)
        F[n - 1] = pref * sinf((float)n * PI_F * u);
}

// ------------------------------------------- per-node e0, charge q, self energy
__global__ void k_e0_q(const float* __restrict__ attrs,
                       const float* __restrict__ ae,
                       const float* __restrict__ fq,
                       const int* __restrict__ batch,
                       float* __restrict__ q,
                       float* __restrict__ out) {
    int n = blockIdx.x * blockDim.x + threadIdx.x;
    if (n >= NN) return;
    float e0 = 0.0f, qv = 0.0f;
    for (int t = 0; t < NEL; ++t) {
        float a = attrs[n * NEL + t];
        e0 += a * ae[t];
        qv += a * fq[t];
    }
    q[n] = qv;
    atomicAdd(&out[batch[n]], e0 + qv * qv * 0.28209479177387814f);
}

// ---------------------------------------------------------------- radial MLP (WMMA)
__device__ __forceinline__ void mlp_stage64(const _Float16* __restrict__ in, // LDS [16][64]
                                            const _Float16* __restrict__ Wp, // packed B
                                            _Float16* __restrict__ outLds,
                                            float* __restrict__ outG,
                                            int lane, bool applySilu) {
    const int  col = lane & 15;
    const bool hi  = lane >= 16;
    const v16h* B = (const v16h*)Wp;
#pragma unroll
    for (int nt = 0; nt < 4; ++nt) {
        v8f acc;
#pragma unroll
        for (int i = 0; i < 8; ++i) acc[i] = 0.0f;
#pragma unroll
        for (int kc = 0; kc < 2; ++kc) {
            v16h a = ldsA(in, col, hi, kc);
            v16h b = B[(kc * 4 + nt) * 32 + lane];
            acc = wmma_f16(a, b, acc);
        }
#pragma unroll
        for (int j = 0; j < 8; ++j) {
            int   M = (hi ? 8 : 0) + j;
            float v = acc[j];
            if (applySilu) v = siluf(v);
            if (outLds) outLds[M * 64 + nt * 16 + col] = (_Float16)v;
            else        outG[M * 64 + nt * 16 + col]   = v;
        }
    }
}

__global__ __launch_bounds__(32)
void k_radial_mlp(const float* __restrict__ ef,       // [E,8]
                  const _Float16* __restrict__ pW0,   // packed stage0 B
                  const _Float16* __restrict__ pW1,
                  const _Float16* __restrict__ pW2,
                  const _Float16* __restrict__ pW3,
                  float* __restrict__ wout) {         // [E,64]
    __shared__ _Float16 bufA[16 * 64];
    __shared__ _Float16 bufB[16 * 64];
    const int  lane = threadIdx.x;
    const int  e0   = blockIdx.x * 16;   // E is a multiple of 16
    const int  col  = lane & 15;
    const bool hi   = lane >= 16;

    // stage 0: unconditional vector loads of the ef row, value-side select
    v4f r0 = *(const v4f*)(ef + (size_t)(e0 + col) * 8);
    v4f r1 = *(const v4f*)(ef + (size_t)(e0 + col) * 8 + 4);
    v16h a;
#pragma unroll
    for (int e2 = 0; e2 < 16; ++e2) {
        float v = (e2 < 8) ? ((e2 < 4) ? r0[e2 & 3] : r1[e2 & 3]) : 0.0f;
        a[e2] = (!hi && e2 < 8) ? (_Float16)v : (_Float16)0.0f;
    }
    const v16h* B0 = (const v16h*)pW0;
#pragma unroll
    for (int nt = 0; nt < 4; ++nt) {
        v16h b = B0[nt * 32 + lane];
        v8f acc;
#pragma unroll
        for (int i = 0; i < 8; ++i) acc[i] = 0.0f;
        acc = wmma_f16(a, b, acc);
#pragma unroll
        for (int j = 0; j < 8; ++j) {
            int M = (hi ? 8 : 0) + j;
            bufA[M * 64 + nt * 16 + col] = (_Float16)siluf(acc[j]);
        }
    }
    __syncthreads();
    mlp_stage64(bufA, pW1, bufB, nullptr, lane, true);
    __syncthreads();
    mlp_stage64(bufB, pW2, bufA, nullptr, lane, true);
    __syncthreads();
    mlp_stage64(bufA, pW3, nullptr, wout + (size_t)e0 * 64, lane, false);
}

// ------------------------------------------------------------ message scatter
__global__ void k_message(const float* __restrict__ w,     // [E,64]
                          const float* __restrict__ feats, // [N,64,16]
                          const float* __restrict__ sh,    // [E,16]
                          const int* __restrict__ eidx,
                          float* __restrict__ agg) {       // [N,64,16]
    int idx = blockIdx.x * blockDim.x + threadIdx.x;       // e*64 + c
    if (idx >= EE * CC) return;
    int c = idx & 63;
    int e = idx >> 6;
    int snd = eidx[e], rcv = eidx[EE + e];
    float m = w[(size_t)e * 64 + c] * feats[(size_t)snd * 1024 + c * 16];
    const float* Y = sh + (size_t)e * 16;
    float* dst = agg + (size_t)rcv * 1024 + c * 16;
#pragma unroll
    for (int k = 0; k < LSH; ++k)
        atomicAdd(&dst[k], m * Y[k]);
}

// ----------------------------------------- node update + readout (WMMA over c)
// out[d,k] = sum_c P[c,k]*Wmix[c,d] + F[c,k]*Wsc[c,d]
// A = packed W^T fragments (global, f16), B = per-node tiles staged TRANSPOSED
// in LDS ([k][c]) so each lane's fragment is one contiguous 32B v16h load.
__global__ __launch_bounds__(32)
void k_node_update(const float* __restrict__ feats,     // [N,64,16]
                   const float* __restrict__ agg,       // [N,64,16] (unscaled)
                   const float* __restrict__ attrs,
                   const float* __restrict__ Wp,        // [NE,3]
                   const _Float16* __restrict__ pWmix,  // packed A
                   const _Float16* __restrict__ pWsc,   // packed A
                   const float* __restrict__ wread0,
                   const float* __restrict__ Wr1,       // [64,16]
                   const float* __restrict__ wr2,       // [16]
                   const int* __restrict__ batch,
                   float* __restrict__ featsOut,        // [N,64,16]
                   float* __restrict__ out,
                   int last) {
    __shared__ _Float16 PhT[16 * 64];   // [k][c]  pf(c) * agg_scaled(c,k)
    __shared__ _Float16 FhT[16 * 64];   // [k][c]  feats(c,k)
    __shared__ float aS[3];
    __shared__ float f0[64];
    __shared__ float hS[16];

    const int  n    = blockIdx.x;
    const int  lane = threadIdx.x;
    const int  col  = lane & 15;
    const bool hi   = lane >= 16;
    const float inv_avg = 1.0f / 12.5f;

    if (lane < 3) {
        float acc = 0.0f;
        for (int t = 0; t < NEL; ++t) acc += attrs[n * NEL + t] * Wp[t * 3 + lane];
        aS[lane] = acc;
    }
    __syncthreads();

    // stage transposed B tiles: each lane handles channels c = lane, lane+32
#pragma unroll
    for (int h = 0; h < 2; ++h) {
        int cc = lane + h * 32;
        const float* ag = agg   + (size_t)n * 1024 + cc * 16;
        const float* fe = feats + (size_t)n * 1024 + cc * 16;
        float s0 = ag[0] * inv_avg;
        float p  = aS[0] + aS[1] * s0 + aS[2] * s0 * s0;
#pragma unroll
        for (int k = 0; k < LSH; ++k) {
            PhT[k * 64 + cc] = (_Float16)(p * (ag[k] * inv_avg));
            FhT[k * 64 + cc] = (_Float16)fe[k];
        }
    }
    __syncthreads();

    const v16h* Amix = (const v16h*)pWmix;
    const v16h* Asc  = (const v16h*)pWsc;
#pragma unroll
    for (int dt = 0; dt < 4; ++dt) {
        v8f acc;
#pragma unroll
        for (int i = 0; i < 8; ++i) acc[i] = 0.0f;
#pragma unroll
        for (int kc = 0; kc < 2; ++kc) {
            v16h aM = Amix[(dt * 2 + kc) * 32 + lane];
            v16h aC = Asc [(dt * 2 + kc) * 32 + lane];
            v16h bP = *(const v16h*)(PhT + col * 64 + kc * 32 + (hi ? 16 : 0));
            v16h bF = *(const v16h*)(FhT + col * 64 + kc * 32 + (hi ? 16 : 0));
            acc = wmma_f16(aM, bP, acc);
            acc = wmma_f16(aC, bF, acc);
        }
#pragma unroll
        for (int j = 0; j < 8; ++j) {
            int   d = dt * 16 + (hi ? 8 : 0) + j;
            float v = acc[j];
            featsOut[(size_t)n * 1024 + d * 16 + col] = v;
            if (col == 0) f0[d] = v;   // lanes 0 and 16 cover all rows
        }
    }
    __syncthreads();

    if (last && lane < HR) {
        float acc = 0.0f;
        for (int c = 0; c < CC; ++c) acc += f0[c] * Wr1[c * HR + lane];
        hS[lane] = siluf(acc);
    }
    __syncthreads();
    if (lane == 0) {
        float ne = 0.0f;
        if (last) { for (int j = 0; j < HR; ++j) ne += hS[j] * wr2[j]; }
        else      { for (int c = 0; c < CC; ++c) ne += f0[c] * wread0[c]; }
        atomicAdd(&out[batch[n]], ne);
    }
}

// ------------------------------------------------------------------ Coulomb (batched)
__global__ void k_coulomb(const float* __restrict__ pos,
                          const float* __restrict__ q,
                          const int* __restrict__ batch,
                          float* __restrict__ out) {
    int i = blockIdx.x * blockDim.x + threadIdx.x;
    if (i >= NN) return;
    int b  = batch[i];
    int j0 = b * NPG, j1 = j0 + NPG;
    float xi = pos[i * 3 + 0], yi = pos[i * 3 + 1], zi = pos[i * 3 + 2];
    float acc = 0.0f;
    for (int j = j0; j < j1; ++j) {
        __builtin_prefetch(&pos[(j + 16) * 3], 0, 0);   // global_prefetch_b8
        if (j == i) continue;
        float dx = xi - pos[j * 3 + 0];
        float dy = yi - pos[j * 3 + 1];
        float dz = zi - pos[j * 3 + 2];
        float r2 = fmaxf(dx * dx + dy * dy + dz * dz, 1e-12f);
        float rr = sqrtf(r2);
        acc += erff(rr * 0.5f) / rr * q[j];   // sigma = 1
    }
    atomicAdd(&out[b], 0.5f * q[i] * acc);
}

// ======================================================================= host
extern "C" void kernel_launch(void* const* d_in, const int* in_sizes, int n_in,
                              void* d_out, int out_size, void* d_ws, size_t ws_size,
                              hipStream_t stream) {
    const float* positions = (const float*)d_in[0];
    const float* node_attrs= (const float*)d_in[1];
    const float* shifts    = (const float*)d_in[2];
    const float* atomic_e  = (const float*)d_in[3];
    const float* formal_q  = (const float*)d_in[4];
    const float* W_embed   = (const float*)d_in[5];
    const float* rW0       = (const float*)d_in[6];   // [2,8,64]
    const float* rW1       = (const float*)d_in[7];   // [2,64,64]
    const float* rW2       = (const float*)d_in[8];
    const float* rW3       = (const float*)d_in[9];
    const float* W_sc      = (const float*)d_in[10];  // [2,64,64]
    const float* W_prod    = (const float*)d_in[11];  // [2,10,3]
    const float* Wmix      = (const float*)d_in[12];  // [2,64,64]
    const float* w_read0   = (const float*)d_in[13];
    const float* W_r1      = (const float*)d_in[14];
    const float* w_r2      = (const float*)d_in[15];
    const int*   edge_index= (const int*)d_in[16];
    const int*   batch     = (const int*)d_in[17];
    float* out = (float*)d_out;

    // workspace carve-up
    float* ws = (float*)d_ws;
    size_t off = 0;
    float* sh     = ws + off; off += (size_t)EE * 16;
    float* ef     = ws + off; off += (size_t)EE * 8;
    float* wEdge  = ws + off; off += (size_t)EE * 64;
    float* qn     = ws + off; off += (size_t)NN;
    float* featsA = ws + off; off += (size_t)NN * 1024;
    float* featsB = ws + off; off += (size_t)NN * 1024;
    float* agg    = ws + off; off += (size_t)NN * 1024;
    _Float16* hws = (_Float16*)(ws + off);
    _Float16* pW0   = hws; hws += 2 * 2048;   // [l][4*32*16]
    _Float16* pW1   = hws; hws += 2 * 4096;   // [l][2*4*32*16]
    _Float16* pW2   = hws; hws += 2 * 4096;
    _Float16* pW3   = hws; hws += 2 * 4096;
    _Float16* pWmix = hws; hws += 2 * 4096;   // A layout
    _Float16* pWsc  = hws; hws += 2 * 4096;   // A layout

    k_zero_out<<<1, 32, 0, stream>>>(out);
    for (int l = 0; l < 2; ++l) {
        k_pack_b0<<<8, 256, 0, stream>>>(rW0 + (size_t)l * 512,  pW0 + (size_t)l * 2048);
        k_pack_b <<<16, 256, 0, stream>>>(rW1 + (size_t)l * 4096, pW1 + (size_t)l * 4096);
        k_pack_b <<<16, 256, 0, stream>>>(rW2 + (size_t)l * 4096, pW2 + (size_t)l * 4096);
        k_pack_b <<<16, 256, 0, stream>>>(rW3 + (size_t)l * 4096, pW3 + (size_t)l * 4096);
        k_pack_a <<<16, 256, 0, stream>>>(Wmix + (size_t)l * 4096, pWmix + (size_t)l * 4096);
        k_pack_a <<<16, 256, 0, stream>>>(W_sc + (size_t)l * 4096, pWsc + (size_t)l * 4096);
    }
    k_feats_init<<<(NN * 1024 + 255) / 256, 256, 0, stream>>>(node_attrs, W_embed, featsA);
    k_edge_geom<<<(EE + 255) / 256, 256, 0, stream>>>(positions, shifts, edge_index, sh, ef);
    k_e0_q<<<(NN + 255) / 256, 256, 0, stream>>>(node_attrs, atomic_e, formal_q, batch, qn, out);

    float* fin  = featsA;
    float* fout = featsB;
    for (int l = 0; l < 2; ++l) {
        k_radial_mlp<<<EE / 16, 32, 0, stream>>>(
            ef, pW0 + (size_t)l * 2048, pW1 + (size_t)l * 4096,
            pW2 + (size_t)l * 4096, pW3 + (size_t)l * 4096, wEdge);
        hipMemsetAsync(agg, 0, (size_t)NN * 1024 * sizeof(float), stream);
        k_message<<<(EE * 64 + 255) / 256, 256, 0, stream>>>(wEdge, fin, sh, edge_index, agg);
        k_node_update<<<NN, 32, 0, stream>>>(
            fin, agg, node_attrs, W_prod + (size_t)l * NEL * 3,
            pWmix + (size_t)l * 4096, pWsc + (size_t)l * 4096,
            w_read0, W_r1, w_r2, batch, fout, out, (l == 1) ? 1 : 0);
        float* t = fin; fin = fout; fout = t;
    }

    k_coulomb<<<(NN + 255) / 256, 256, 0, stream>>>(positions, qn, batch, out);
}
